// MultiHeadAttention_66228395704755
// MI455X (gfx1250) — matrix operand order repaired and checked by reading, not compiled, not run on hardware
//
#include <hip/hip_runtime.h>

typedef __attribute__((ext_vector_type(16))) __bf16 v16bf;
typedef __attribute__((ext_vector_type(8)))  __bf16 bf16x8;
typedef __attribute__((ext_vector_type(2)))  __bf16 bf16x2;
typedef __attribute__((ext_vector_type(8)))  float  v8f;

#define WMMA_BF16(a, b, c) \
  __builtin_amdgcn_wmma_f32_16x16x32_bf16(false, (a), false, (b), (short)0, (c), false, false)

#define B_SZ   2
#define NSEQ   2048
#define DMODEL 1024
#define HEADS  16
#define DK     64
#define QKV3   192   // 3*DK per head, interleaved (B,N,H,192) layout

__device__ __forceinline__ v16bf mk16(bf16x8 lo, bf16x8 hi) {
  v16bf r;
#pragma unroll
  for (int i = 0; i < 8; ++i) { r[i] = lo[i]; r[i + 8] = hi[i]; }
  return r;
}

// ---------------------------------------------------------------------------
// fp32 -> bf16 convert (grid-stride) and zero-fill
// ---------------------------------------------------------------------------
__global__ void cvt_f32_bf16(const float* __restrict__ src, __bf16* __restrict__ dst, int n) {
  for (int i = blockIdx.x * blockDim.x + threadIdx.x; i < n; i += gridDim.x * blockDim.x)
    dst[i] = (__bf16)src[i];
}

__global__ void zero_f32(float* __restrict__ p, int n) {
  int i = blockIdx.x * blockDim.x + threadIdx.x;
  if (i < n) p[i] = 0.0f;
}

// ---------------------------------------------------------------------------
// bf16 WMMA GEMM: C[M,N] = A[M,K] @ B[K,N] + bias
// 64x64 tile / 128 threads (4 waves), K-step 32, double-buffered LDS.
// outF != nullptr -> f32 output, else bf16 output to outB.
// ---------------------------------------------------------------------------
__global__ __launch_bounds__(128) void gemm_bf16_wmma(
    const __bf16* __restrict__ A, const __bf16* __restrict__ Bm,
    const float* __restrict__ bias,
    float* __restrict__ outF, __bf16* __restrict__ outB,
    int M, int N, int K)
{
  __shared__ __bf16 Asl[2][64 * 32];   // A tile, row-major (row stride 32)
  __shared__ __bf16 BTsl[2][64 * 32];  // B tile transposed: [n][k]

  const int n0   = blockIdx.x * 64;
  const int m0   = blockIdx.y * 64;
  const int tid  = threadIdx.x;
  const int wave = tid >> 5;
  const int lane = tid & 31;
  const int lr   = lane & 15;
  const bool hi  = lane >= 16;

  // staging coordinates
  const int ar  = tid >> 1;            // A row (0..63)
  const int ako = (tid & 1) * 16;      // A k-offset (0/16)
  const int bk  = (tid >> 3) * 2;      // B even k (0..30)
  const int bno = (tid & 7) * 8;       // B n-offset group of 8

  v8f acc[4] = {};

  auto loadG = [&](int kt, bf16x8& a0, bf16x8& a1, bf16x8& b0, bf16x8& b1) {
    const __bf16* ap = A + (size_t)(m0 + ar) * K + kt + ako;
    a0 = *(const bf16x8*)(ap);
    a1 = *(const bf16x8*)(ap + 8);
    const __bf16* bp = Bm + (size_t)(kt + bk) * N + n0 + bno;
    b0 = *(const bf16x8*)(bp);
    b1 = *(const bf16x8*)(bp + N);
  };
  auto storeL = [&](int buf, bf16x8 a0, bf16x8 a1, bf16x8 b0, bf16x8 b1) {
    *(bf16x8*)(Asl[buf] + ar * 32 + ako)     = a0;
    *(bf16x8*)(Asl[buf] + ar * 32 + ako + 8) = a1;
#pragma unroll
    for (int i = 0; i < 8; ++i) {       // packed pair store: ds_store_b32
      bf16x2 p; p[0] = b0[i]; p[1] = b1[i];
      *(bf16x2*)(BTsl[buf] + (bno + i) * 32 + bk) = p;
    }
  };
  auto compute = [&](int buf) {
    const __bf16* ap = Asl[buf] + (wave * 16 + lr) * 32 + (hi ? 8 : 0);
    v16bf afrag = mk16(*(const bf16x8*)ap, *(const bf16x8*)(ap + 16));
    v16bf bfr[4];
#pragma unroll
    for (int ct = 0; ct < 4; ++ct) {    // load all frags, then chain WMMAs
      const __bf16* bp = BTsl[buf] + (ct * 16 + lr) * 32 + (hi ? 8 : 0);
      bfr[ct] = mk16(*(const bf16x8*)bp, *(const bf16x8*)(bp + 16));
    }
#pragma unroll
    for (int ct = 0; ct < 4; ++ct)
      acc[ct] = WMMA_BF16(afrag, bfr[ct], acc[ct]);
  };

  const int nk = K >> 5;
  {
    bf16x8 a0, a1, b0, b1;
    loadG(0, a0, a1, b0, b1);
    storeL(0, a0, a1, b0, b1);
  }
  __syncthreads();
  for (int t = 0; t < nk; ++t) {
    const int cur = t & 1;
    bf16x8 a0, a1, b0, b1;
    if (t + 1 < nk) {
      loadG((t + 1) * 32, a0, a1, b0, b1);
      if (t + 2 < nk) {                 // stream-ahead hint: global_prefetch_b8
        __builtin_prefetch(A + (size_t)(m0 + ar) * K + (t + 2) * 32 + ako, 0, 0);
        __builtin_prefetch(Bm + (size_t)((t + 2) * 32 + bk) * N + n0 + bno, 0, 0);
      }
    }
    compute(cur);
    if (t + 1 < nk) {
      storeL(cur ^ 1, a0, a1, b0, b1);
      __syncthreads();
    }
  }

  // epilogue: bias + store (C layout: VGPR j -> row j + (hi?8:0), col = lane%16)
#pragma unroll
  for (int ct = 0; ct < 4; ++ct) {
    int col = n0 + ct * 16 + lr;
    float bv = bias ? bias[col] : 0.0f;
    int rbase = m0 + wave * 16 + (hi ? 8 : 0);
#pragma unroll
    for (int j = 0; j < 8; ++j) {
      float v = acc[ct][j] + bv;
      if (outF) outF[(size_t)(rbase + j) * N + col] = v;
      else      outB[(size_t)(rbase + j) * N + col] = (__bf16)v;
    }
  }
}

// ---------------------------------------------------------------------------
// Attention: one block per (b, h, 16-query tile). 128 threads / 4 waves.
// Dynamic LDS: S[16][2048] f32 raw-exp (128KB) + VT[64][128] bf16 (16KB)
//            + red[128] f32 + rowinv[16] f32
// Normalization folded into attn_mean sums and the context epilogue.
// ---------------------------------------------------------------------------
__global__ __launch_bounds__(128) void attn_wmma(
    const __bf16* __restrict__ qkv,   // (B, N, H, 192) bf16
    __bf16* __restrict__ ctx,         // (B*N, 1024) bf16
    float* __restrict__ attn_mean)    // (B, N) f32 (pre-zeroed)
{
  extern __shared__ char smem[];
  float*  S      = (float*)smem;                                  // 16 * 2048
  __bf16* VT     = (__bf16*)(smem + 16 * NSEQ * 4);               // 64 * 128
  float*  red    = (float*)(smem + 16 * NSEQ * 4 + 64 * 128 * 2); // 128
  float*  rowinv = red + 128;                                     // 16

  const int wg = blockIdx.x;
  const int qt = wg & 127;
  const int h  = (wg >> 7) & 15;
  const int b  = wg >> 11;
  const int q0 = qt * 16;

  const int tid  = threadIdx.x;
  const int wave = tid >> 5;
  const int lane = tid & 31;
  const int lr   = lane & 15;
  const bool hi  = lane >= 16;

  const __bf16* qkv_b = qkv + (size_t)b * NSEQ * HEADS * QKV3;

  // Q fragments for this query tile, kept in registers (reused 128x)
  v16bf qfrag[2];
  {
    const __bf16* qrow = qkv_b + ((size_t)(q0 + lr) * HEADS + h) * QKV3;
#pragma unroll
    for (int s = 0; s < 2; ++s) {
      int dbase = s * 32 + (hi ? 8 : 0);
      qfrag[s] = mk16(*(const bf16x8*)(qrow + dbase),
                      *(const bf16x8*)(qrow + dbase + 16));
    }
  }

  // ---- Pass 1: S = Q @ K^T (keys tiled by 16, waves interleaved) ----
  for (int kt = wave; kt < NSEQ / 16; kt += 4) {
    const __bf16* krow = qkv_b + ((size_t)(kt * 16 + lr) * HEADS + h) * QKV3 + DK;
    if (kt + 4 < NSEQ / 16)
      __builtin_prefetch(krow + (size_t)64 * HEADS * QKV3, 0, 0);
    v8f acc = {};
#pragma unroll
    for (int s = 0; s < 2; ++s) {
      int dbase = s * 32 + (hi ? 8 : 0);
      v16bf bfrag = mk16(*(const bf16x8*)(krow + dbase),
                         *(const bf16x8*)(krow + dbase + 16));
      acc = WMMA_BF16(qfrag[s], bfrag, acc);
    }
    int col = kt * 16 + lr;
    int rbase = hi ? 8 : 0;
#pragma unroll
    for (int j = 0; j < 8; ++j) S[(rbase + j) * NSEQ + col] = acc[j];
  }
  __syncthreads();

  // ---- Pass 2: row max + exp + row-sum (8 threads per row); keep raw exp ----
  const float scale = 0.125f;  // 1/sqrt(64)
  {
    int row = tid >> 3, sub = tid & 7;
    float* srow = S + row * NSEQ + sub * 256;
    float m = -1e30f;
    for (int i = 0; i < 256; ++i) m = fmaxf(m, srow[i]);
    red[tid] = m;
    __syncthreads();
    float rm = -1e30f;
#pragma unroll
    for (int i = 0; i < 8; ++i) rm = fmaxf(rm, red[row * 8 + i]);
    __syncthreads();
    float sum = 0.0f;
    for (int i = 0; i < 256; ++i) {
      float p = __expf((srow[i] - rm) * scale);
      srow[i] = p;
      sum += p;
    }
    red[tid] = sum;
    __syncthreads();
    float rs = 0.0f;
#pragma unroll
    for (int i = 0; i < 8; ++i) rs += red[row * 8 + i];
    if (sub == 0) rowinv[row] = 1.0f / rs;
  }
  __syncthreads();

  // ---- attn_mean[b, m] += (1/(H*N)) * sum_r exp[r][m] * rowinv[r] ----
  {
    float* am = attn_mean + (size_t)b * NSEQ;
    const float w = 1.0f / (float)(HEADS * NSEQ);
#pragma unroll
    for (int i = 0; i < 16; ++i) {
      int mcol = i * 128 + tid;
      float cs = 0.0f;
#pragma unroll
      for (int r = 0; r < 16; ++r) cs += S[r * NSEQ + mcol] * rowinv[r];
      atomicAdd(&am[mcol], cs * w);
    }
  }

  // ---- Pass 3: context = (exp) @ V, row-scaled at the end.
  //      Wave ct owns dk columns [ct*16, ct*16+16) ----
  v8f cacc = {};
  const int ct = wave;
  const int vp    = tid >> 1;          // key pair index (0..63)
  const int dhalf = (tid & 1) * 32;    // d range half
  for (int mt = 0; mt < 16; ++mt) {    // macro key blocks of 128
    __syncthreads();
    {   // stage VT[d][key] for 128 keys, packed-pair ds_store_b32
      int key0 = mt * 128 + vp * 2;
      const __bf16* vr0 = qkv_b + ((size_t)key0 * HEADS + h) * QKV3 + 2 * DK + dhalf;
      const __bf16* vr1 = vr0 + (size_t)HEADS * QKV3;
      __bf16 t0[32], t1[32];
#pragma unroll
      for (int i = 0; i < 4; ++i) {
        *(bf16x8*)(t0 + i * 8) = *(const bf16x8*)(vr0 + i * 8);
        *(bf16x8*)(t1 + i * 8) = *(const bf16x8*)(vr1 + i * 8);
      }
#pragma unroll
      for (int d = 0; d < 32; ++d) {
        bf16x2 p; p[0] = t0[d]; p[1] = t1[d];
        *(bf16x2*)(VT + (dhalf + d) * 128 + vp * 2) = p;
      }
    }
    __syncthreads();
#pragma unroll
    for (int ks = 0; ks < 4; ++ks) {  // K-steps of 32 within block
      int kk = ks * 32;
      // A fragment: raw exp from LDS, f32 -> bf16
      v16bf afrag;
      {
        const float* prow = S + lr * NSEQ + mt * 128 + kk + (hi ? 8 : 0);
        float4 f0 = *(const float4*)(prow);
        float4 f1 = *(const float4*)(prow + 4);
        float4 f2 = *(const float4*)(prow + 16);
        float4 f3 = *(const float4*)(prow + 20);
        afrag[0]  = (__bf16)f0.x; afrag[1]  = (__bf16)f0.y;
        afrag[2]  = (__bf16)f0.z; afrag[3]  = (__bf16)f0.w;
        afrag[4]  = (__bf16)f1.x; afrag[5]  = (__bf16)f1.y;
        afrag[6]  = (__bf16)f1.z; afrag[7]  = (__bf16)f1.w;
        afrag[8]  = (__bf16)f2.x; afrag[9]  = (__bf16)f2.y;
        afrag[10] = (__bf16)f2.z; afrag[11] = (__bf16)f2.w;
        afrag[12] = (__bf16)f3.x; afrag[13] = (__bf16)f3.y;
        afrag[14] = (__bf16)f3.z; afrag[15] = (__bf16)f3.w;
      }
      // B fragment: V^T tile from LDS (two 8-runs per lane, table order)
      const __bf16* vt = VT + (ct * 16 + lr) * 128 + kk + (hi ? 8 : 0);
      v16bf bfrag = mk16(*(const bf16x8*)vt, *(const bf16x8*)(vt + 16));
      cacc = WMMA_BF16(afrag, bfrag, cacc);
    }
  }
  // store context tile (bf16), applying softmax row normalization here
  {
    int colbase = h * DK + ct * 16 + lr;
    int rbase = hi ? 8 : 0;
#pragma unroll
    for (int j = 0; j < 8; ++j) {
      float v = cacc[j] * rowinv[rbase + j];
      ctx[((size_t)b * NSEQ + q0 + rbase + j) * DMODEL + colbase] = (__bf16)v;
    }
  }
}

// ---------------------------------------------------------------------------
extern "C" void kernel_launch(void* const* d_in, const int* in_sizes, int n_in,
                              void* d_out, int out_size, void* d_ws, size_t ws_size,
                              hipStream_t stream) {
  (void)in_sizes; (void)n_in; (void)out_size; (void)ws_size;

  const float* x     = (const float*)d_in[0];   // (2, 2048, 1024)
  const float* w_qkv = (const float*)d_in[1];   // (1024, 3072)
  const float* b_qkv = (const float*)d_in[2];   // (3072,)
  const float* w_out = (const float*)d_in[3];   // (1024, 1024)
  const float* b_out = (const float*)d_in[4];   // (1024,)

  float* out       = (float*)d_out;                          // (2,2048,1024)
  float* attn_mean = out + (size_t)B_SZ * NSEQ * DMODEL;     // (2,2048)

  // workspace carve-up (bf16 buffers)
  char* ws = (char*)d_ws;
  __bf16* x_bf    = (__bf16*)(ws);                 //  8 MiB: 4096x1024
  __bf16* wqkv_bf = (__bf16*)(ws + 8388608);       //  6 MiB: 1024x3072
  __bf16* wout_bf = (__bf16*)(ws + 14680064);      //  2 MiB: 1024x1024
  __bf16* qkv_bf  = (__bf16*)(ws + 16777216);      // 24 MiB: 4096x3072
  __bf16* ctx_bf  = (__bf16*)(ws + 41943040);      //  8 MiB: 4096x1024

  // 0) converts + zero attn_mean accumulator
  cvt_f32_bf16<<<2048, 256, 0, stream>>>(x,     x_bf,    B_SZ * NSEQ * DMODEL);
  cvt_f32_bf16<<<2048, 256, 0, stream>>>(w_qkv, wqkv_bf, DMODEL * 3 * DMODEL);
  cvt_f32_bf16<<<1024, 256, 0, stream>>>(w_out, wout_bf, DMODEL * DMODEL);
  zero_f32<<<16, 256, 0, stream>>>(attn_mean, B_SZ * NSEQ);

  // 1) qkv = x @ w_qkv + b_qkv  -> bf16, (B*N, 3072) == (B,N,H,192) interleaved
  gemm_bf16_wmma<<<dim3(3 * DMODEL / 64, B_SZ * NSEQ / 64), 128, 0, stream>>>(
      x_bf, wqkv_bf, b_qkv, nullptr, qkv_bf, B_SZ * NSEQ, 3 * DMODEL, DMODEL);

  // 2) attention (scores in LDS, softmax, context, attn_mean atomics)
  size_t lds_bytes = (size_t)16 * NSEQ * 4 + (size_t)64 * 128 * 2 + 128 * 4 + 64;
  attn_wmma<<<B_SZ * HEADS * (NSEQ / 16), 128, lds_bytes, stream>>>(
      qkv_bf, ctx_bf, attn_mean);

  // 3) out = context @ w_out + b_out  -> f32
  gemm_bf16_wmma<<<dim3(DMODEL / 64, B_SZ * NSEQ / 64), 128, 0, stream>>>(
      ctx_bf, wout_bf, b_out, out, nullptr, B_SZ * NSEQ, DMODEL, DMODEL);
}
